// NPairLoss_85487029060330
// MI455X (gfx1250) — matrix operand order repaired and checked by reading, not compiled, not run on hardware
//
#include <hip/hip_runtime.h>
#include <hip/hip_bf16.h>
#include <math.h>

// N-pair contrastive loss, fused for MI455X (gfx1250, wave32, WMMA bf16).
//   1) zero scalar accumulator
//   2) prep: fp32 -> bf16 hi/lo split planes, row norms, pos_sim, diag,
//      rowsum init = exp(pos-1), atomic global sum-of-squares (L2 term)
//   3) fused GEMM (bf16x3 split WMMA); B tile staged in LDS via
//      global_load_async_to_lds_b128 (ASYNCcnt) + streaming softmax denom
//   4) finalize: loss = mean(log(rowsum) - pos + 1) + 0.02*sqrt(sumsq)

#define NROWS 8192
#define DIM   512
#define EPS   1e-8f
#define L2REG 0.02f

#define KCHUNK  128              // K elements staged per LDS refill
#define BSTRIDE 136              // 128 + 8 pad elems: 272B rows (16B aligned,
                                 // bank step 4 -> uniform 2-phase b128 reads)

typedef __attribute__((ext_vector_type(16))) __bf16 v16bf;
typedef __attribute__((ext_vector_type(8)))  float  v8f;

__device__ __forceinline__ unsigned short f2bf(float f) {
    unsigned int u = __float_as_uint(f);
    u += 0x7fffu + ((u >> 16) & 1u);          // round-to-nearest-even
    return (unsigned short)(u >> 16);
}
__device__ __forceinline__ float bf2f(unsigned short h) {
    return __uint_as_float(((unsigned int)h) << 16);
}

// Async global -> LDS 16B copy (gfx1250, tracked by ASYNCcnt).
__device__ __forceinline__ void async_b128(unsigned lds_off, unsigned long long gaddr) {
    asm volatile("global_load_async_to_lds_b128 %0, %1, off"
                 :: "v"(lds_off), "v"(gaddr) : "memory");
}
__device__ __forceinline__ void wait_async0() {
    asm volatile("s_wait_asynccnt 0x0" ::: "memory");
}

__global__ void npair_zero_kernel(float* sumsq) {
    if (threadIdx.x == 0) *sumsq = 0.0f;
}

// One block (256 threads) per example row: norms, pos_sim, diag, bf16 split.
__global__ __launch_bounds__(256)
void npair_prep_kernel(const float* __restrict__ ex,
                       unsigned short* __restrict__ Ah, unsigned short* __restrict__ Al,
                       unsigned short* __restrict__ Ph, unsigned short* __restrict__ Pl,
                       float* __restrict__ a_norm, float* __restrict__ p_norm,
                       float* __restrict__ pos,    float* __restrict__ diagv,
                       float* __restrict__ rowsum, float* __restrict__ sumsq) {
    __shared__ float sA[256], sP[256], sD[256];
    const int r = blockIdx.x, tid = threadIdx.x;
    const float* arow = ex + (size_t)r * (2 * DIM);
    const float* prow = arow + DIM;
    float asq = 0.f, psq = 0.f, dot = 0.f;
    for (int c = tid; c < DIM; c += 256) {
        float a = arow[c], p = prow[c];
        asq += a * a; psq += p * p; dot += a * p;
        unsigned short ahv = f2bf(a);
        unsigned short alv = f2bf(a - bf2f(ahv));
        unsigned short phv = f2bf(p);
        unsigned short plv = f2bf(p - bf2f(phv));
        size_t o = (size_t)r * DIM + c;
        Ah[o] = ahv; Al[o] = alv; Ph[o] = phv; Pl[o] = plv;
    }
    sA[tid] = asq; sP[tid] = psq; sD[tid] = dot;
    __syncthreads();
    for (int s = 128; s > 0; s >>= 1) {
        if (tid < s) { sA[tid] += sA[tid+s]; sP[tid] += sP[tid+s]; sD[tid] += sD[tid+s]; }
        __syncthreads();
    }
    if (tid == 0) {
        float asq_t = sA[0], psq_t = sP[0], dot_t = sD[0];
        float an = sqrtf(asq_t), pn = sqrtf(psq_t);
        a_norm[r] = an; p_norm[r] = pn;
        float ps = dot_t / fmaxf(an * pn, EPS);
        pos[r]   = ps;
        diagv[r] = asq_t / fmaxf(asq_t, EPS);
        rowsum[r] = __expf(ps - 1.0f);            // pos-logit term, shift = 1.0
        atomicAdd(sumsq, asq_t + psq_t);
    }
}

// Fused bf16x3-split WMMA GEMM + streaming softmax denominator.
// Block: 8 waves; each wave computes a 16(M) x 64(N) strip -> block tile 128x64.
// B tile (64 cols x KCHUNK, hi+lo planes) staged in LDS via async copies.
__global__ __launch_bounds__(256)
void npair_gemm_kernel(const unsigned short* __restrict__ Ah,
                       const unsigned short* __restrict__ Al,
                       const unsigned short* __restrict__ Ph,
                       const unsigned short* __restrict__ Pl,
                       const float* __restrict__ a_norm,
                       const float* __restrict__ p_norm,
                       const float* __restrict__ diagv,
                       float* __restrict__ rowsum) {
    __shared__ unsigned short sBh[64 * BSTRIDE];
    __shared__ unsigned short sBl[64 * BSTRIDE];

    const int tid  = threadIdx.x;
    const int lane = tid & 31;
    const int wave = tid >> 5;
    const int lr   = lane & 15;       // row/col index within 16-wide group
    const int half = lane >> 4;       // K-half selector per ISA VGPR layout
    const int rowBase = blockIdx.y * 128 + wave * 16;
    const int colBase = blockIdx.x * 64;

    union Frag { v16bf v; uint4 q[2]; };

    const size_t aoff = (size_t)(rowBase + lr) * DIM;

    v8f acc[4] = {};
    for (int kc = 0; kc < DIM; kc += KCHUNK) {
        __syncthreads();   // previous chunk fully consumed by all waves
        // Cooperative async fill: 64 rows x (KCHUNK/8) 16B segments per plane,
        // 1024 segments / 256 threads = 4 issues per thread per plane.
        #pragma unroll
        for (int it = 0; it < 4; ++it) {
            const int seg = tid + it * 256;
            const int row = seg >> 4;            // 0..63
            const int ks  = (seg & 15) << 3;     // 0,8,...,120 (elements)
            const size_t g = (size_t)(colBase + row) * DIM + kc + ks;
            const unsigned lds = (unsigned)(row * BSTRIDE + ks) * 2u;
            async_b128((unsigned)(size_t)(void*)sBh + lds, (unsigned long long)(Ph + g));
            async_b128((unsigned)(size_t)(void*)sBl + lds, (unsigned long long)(Pl + g));
        }
        wait_async0();     // drain this wave's async copies
        __syncthreads();   // all waves' staged data visible

        for (int kk = 0; kk < KCHUNK; kk += 32) {
            const int k = kc + kk;
            // A fragment (16-bit A layout): lane<16 -> K {k..k+7, k+16..k+23}
            //                               lane>=16 -> K {k+8..k+15, k+24..k+31}
            const int ka0 = k + half * 8;
            Frag ah, al;
            ah.q[0] = *(const uint4*)(Ah + aoff + ka0);
            ah.q[1] = *(const uint4*)(Ah + aoff + ka0 + 16);
            al.q[0] = *(const uint4*)(Al + aoff + ka0);
            al.q[1] = *(const uint4*)(Al + aoff + ka0 + 16);
            if (k + 32 < DIM)
                __builtin_prefetch((const void*)(Ah + aoff + ka0 + 32), 0, 3);

            #pragma unroll
            for (int nt = 0; nt < 4; ++nt) {
                // B fragment (matches A): lane<16 -> K {kk..kk+7, kk+16..kk+23},
                // lane>=16 -> K {kk+8..kk+15, kk+24..kk+31}; lane holds column n = staged row.
                const int brow = nt * 16 + lr;
                const int koff = kk + half * 8;
                Frag bh, bl;
                bh.q[0] = *(const uint4*)&sBh[brow * BSTRIDE + koff];
                bh.q[1] = *(const uint4*)&sBh[brow * BSTRIDE + koff + 16];
                bl.q[0] = *(const uint4*)&sBl[brow * BSTRIDE + koff];
                bl.q[1] = *(const uint4*)&sBl[brow * BSTRIDE + koff + 16];
                // a*b ~= ah*bh + ah*bl + al*bh  (al*bl ~ 2^-16, dropped)
                acc[nt] = __builtin_amdgcn_wmma_f32_16x16x32_bf16(
                    false, ah.v, false, bh.v, (short)0, acc[nt], false, false);
                acc[nt] = __builtin_amdgcn_wmma_f32_16x16x32_bf16(
                    false, ah.v, false, bl.v, (short)0, acc[nt], false, false);
                acc[nt] = __builtin_amdgcn_wmma_f32_16x16x32_bf16(
                    false, al.v, false, bh.v, (short)0, acc[nt], false, false);
            }
        }
    }

    // Epilogue. C layout: lane<16 VGPR v -> (M=v, N=lr); lane>=16 -> (M=8+v).
    float an[8];
    #pragma unroll
    for (int v = 0; v < 8; ++v) an[v] = a_norm[rowBase + half * 8 + v];

    float esum[8] = {0.f,0.f,0.f,0.f,0.f,0.f,0.f,0.f};
    #pragma unroll
    for (int nt = 0; nt < 4; ++nt) {
        const int n = colBase + nt * 16 + lr;
        const float pn = p_norm[n];
        #pragma unroll
        for (int v = 0; v < 8; ++v) {
            const int m = rowBase + half * 8 + v;
            float sim = acc[nt][v] / fmaxf(an[v] * pn, EPS);
            if (m == n) sim = diagv[m];           // diagonal substitution
            esum[v] += __expf(sim - 1.0f);        // shift by max bound 1.0
        }
    }
    // 16 lanes (same half) share each output row: butterfly-reduce, then
    // one atomic per row per tile-column.
    #pragma unroll
    for (int v = 0; v < 8; ++v) {
        float s = esum[v];
        s += __shfl_xor(s, 1, 16);
        s += __shfl_xor(s, 2, 16);
        s += __shfl_xor(s, 4, 16);
        s += __shfl_xor(s, 8, 16);
        if (lr == 0)
            atomicAdd(&rowsum[rowBase + half * 8 + v], s);
    }
}

__global__ __launch_bounds__(256)
void npair_finalize_kernel(const float* __restrict__ rowsum,
                           const float* __restrict__ pos,
                           const float* __restrict__ sumsq,
                           float* __restrict__ out) {
    __shared__ float s[256];
    const int tid = threadIdx.x;
    float local = 0.f;
    for (int i = tid; i < NROWS; i += 256)
        local += logf(rowsum[i]) - pos[i] + 1.0f;   // -( (pos-1) - log(denom) )
    s[tid] = local;
    __syncthreads();
    for (int sw = 128; sw > 0; sw >>= 1) {
        if (tid < sw) s[tid] += s[tid + sw];
        __syncthreads();
    }
    if (tid == 0)
        out[0] = s[0] / (float)NROWS + L2REG * sqrtf(*sumsq);
}

extern "C" void kernel_launch(void* const* d_in, const int* in_sizes, int n_in,
                              void* d_out, int out_size, void* d_ws, size_t ws_size,
                              hipStream_t stream) {
    const float* ex = (const float*)d_in[0];
    float* out = (float*)d_out;

    // Workspace layout (~33.7 MB): 4 bf16 planes + per-row fp32 vectors.
    char* w = (char*)d_ws;
    const size_t mat = (size_t)NROWS * DIM * sizeof(unsigned short);  // 8 MB
    unsigned short* Ah = (unsigned short*)(w);
    unsigned short* Al = (unsigned short*)(w + 1 * mat);
    unsigned short* Ph = (unsigned short*)(w + 2 * mat);
    unsigned short* Pl = (unsigned short*)(w + 3 * mat);
    float* fbase  = (float*)(w + 4 * mat);
    float* a_norm = fbase + 0 * NROWS;
    float* p_norm = fbase + 1 * NROWS;
    float* pos    = fbase + 2 * NROWS;
    float* diagv  = fbase + 3 * NROWS;
    float* rowsum = fbase + 4 * NROWS;
    float* sumsq  = fbase + 5 * NROWS;

    npair_zero_kernel<<<1, 32, 0, stream>>>(sumsq);
    npair_prep_kernel<<<NROWS, 256, 0, stream>>>(ex, Ah, Al, Ph, Pl,
                                                 a_norm, p_norm, pos, diagv,
                                                 rowsum, sumsq);
    dim3 grid(NROWS / 64, NROWS / 128);
    npair_gemm_kernel<<<grid, 256, 0, stream>>>(Ah, Al, Ph, Pl,
                                                a_norm, p_norm, diagv, rowsum);
    npair_finalize_kernel<<<1, 256, 0, stream>>>(rowsum, pos, sumsq, out);
}
